// CouplingLayer_88579405513047
// MI455X (gfx1250) — compile-verified, matrix operand order checked by reading, not run on hardware
//
#include <hip/hip_runtime.h>
#include <cmath>

#define SP      2048      // padded sequence (2047 valid rows + 1 zero row)
#define SVALID  2047
#define DIMM    256
#define HIDD    512
#define NHEADS  8
#define DH      32
#define EMBK    64        // cond features padded 35 -> 64
#define AFFN    64        // spline params padded 49 -> 64

typedef __attribute__((ext_vector_type(16))) __bf16 v16bf;
typedef __attribute__((ext_vector_type(8)))  float  v8f;
typedef unsigned short ushort_t;

// ---------------- helpers ----------------

__device__ inline ushort_t f2bf(float x) {
  unsigned int u = __float_as_uint(x);
  unsigned int r = u + 0x7FFFu + ((u >> 16) & 1u);   // round-to-nearest-even
  return (ushort_t)(r >> 16);
}

// Load a 16x32 bf16 fragment in the CDNA5 WMMA A/B per-lane layout from a
// row-major [rows x ld] bf16 (ushort) buffer. lane<16: K {0..7,16..23};
// lane>=16: K {8..15,24..31}; row = row0 + (lane&15).
__device__ inline v16bf load_frag(const ushort_t* base, int ld, int row0, int col0) {
  int lane = threadIdx.x & 31;
  int hi   = lane >> 4;
  const __bf16* p = (const __bf16*)base + (size_t)(row0 + (lane & 15)) * ld + col0;
  v16bf f;
#pragma unroll
  for (int i = 0; i < 8; ++i) {
    f[i]     = p[hi * 8 + i];
    f[8 + i] = p[16 + hi * 8 + i];
  }
  return f;
}

// Same fragment pattern out of an LDS tile laid out [rows][32] bf16.
__device__ inline v16bf lds_frag(const ushort_t* tile_row0) {
  int lane = threadIdx.x & 31;
  int hi   = lane >> 4;
  const __bf16* p = (const __bf16*)(tile_row0 + (size_t)(lane & 15) * 32);
  v16bf f;
#pragma unroll
  for (int i = 0; i < 8; ++i) {
    f[i]     = p[hi * 8 + i];
    f[8 + i] = p[16 + hi * 8 + i];
  }
  return f;
}

// Issue one 16-byte gfx1250 async global->LDS transfer (tracked by ASYNCcnt).
__device__ inline void async_copy_b128(const void* gptr, const void* lptr) {
  unsigned long long ga = (unsigned long long)gptr;
  unsigned int       lo = (unsigned int)(unsigned long long)lptr;
  asm volatile("global_load_async_to_lds_b128 %0, %1, off"
               :: "v"(lo), "v"(ga) : "memory");
}

// ---------------- kernels ----------------

// fp32 [Nin x Kin] -> bf16 [Np x Kp] with zero padding (N rows and K cols)
__global__ void convert_pad_kernel(const float* __restrict__ src, ushort_t* __restrict__ dst,
                                   int Nin, int Kin, int Np, int Kp) {
  int i = blockIdx.x * blockDim.x + threadIdx.x;
  int tot = Np * Kp;
  if (i >= tot) return;
  int nn = i / Kp, kk = i % Kp;
  float v = (nn < Nin && kk < Kin) ? src[(size_t)nn * Kin + kk] : 0.0f;
  dst[i] = f2bf(v);
}

// circular features + scalar conditioners -> bf16 [SP x 64] (rows >= 2047 zero)
__global__ void cond_kernel(const float* __restrict__ pos, const float* __restrict__ scale,
                            const float* __restrict__ press, const float* __restrict__ temp,
                            ushort_t* __restrict__ X0) {
  int row = blockIdx.x * blockDim.x + threadIdx.x;
  if (row >= SP) return;
  ushort_t* o = X0 + (size_t)row * EMBK;
  if (row >= SVALID) { for (int j = 0; j < EMBK; ++j) o[j] = 0; return; }
  const float TWO_PI = 6.283185307179586f;
  for (int d = 0; d < 2; ++d) {
    float x = pos[(size_t)(row + 1) * 3 + d];
    for (int f = 0; f < 8; ++f) {
      float ang = x * TWO_PI * (float)(f + 1);
      o[d * 16 + f]     = f2bf(__cosf(ang));
      o[d * 16 + 8 + f] = f2bf(__sinf(ang));
    }
  }
  o[32] = f2bf(*scale);
  o[33] = f2bf(*temp);
  o[34] = f2bf(*press);
  for (int j = 35; j < EMBK; ++j) o[j] = 0;
}

__global__ void transpose_kernel(const ushort_t* __restrict__ src, ushort_t* __restrict__ dst) {
  int idx = blockIdx.x * blockDim.x + threadIdx.x;
  if (idx >= SP * DIMM) return;
  int r = idx / DIMM, c = idx % DIMM;
  dst[(size_t)c * SP + r] = src[idx];
}

// C[M x N] = A[M x K] * B[N x K]^T (+bias, +residual, relu), bf16 WMMA, fp32 acc.
// Block = 128 threads = 4 waves; block computes a 64(M) x 64(N) tile.
// Wave w owns rows m0+16w x all 64 cols: 4 accumulators, A-fragment reused 4x.
// The shared 64x32 B tile is staged into LDS with gfx1250 async loads
// (global_load_async_to_lds_b128, ASYNCcnt) each K-step; both chunks are
// issued unconditionally so the async path runs at full EXEC.
__global__ void gemm_bf16_kernel(const ushort_t* __restrict__ A, const ushort_t* __restrict__ B,
                                 const float* __restrict__ bias, int Nb,
                                 const float* __restrict__ resid,
                                 float* __restrict__ outF, ushort_t* __restrict__ outB,
                                 int N, int K, int relu) {
  __shared__ __align__(16) ushort_t bt[64][32];   // B tile: 64 N-rows x 32 K (4 KB)
  int tid  = threadIdx.x;
  int wid  = tid >> 5;
  int lane = tid & 31;
  int hi   = lane >> 4;
  int m0   = blockIdx.x * 64 + wid * 16;
  int n0   = blockIdx.y * 64;

  // staging chunk assignment: 256 x 16B chunks; thread handles tid and tid+128
  int r0 = tid >> 2,         c0 = (tid & 3) * 8;          // chunk tid
  int r1 = (tid + 128) >> 2, c1 = ((tid + 128) & 3) * 8;  // chunk tid+128

  v8f acc[4];
#pragma unroll
  for (int j = 0; j < 4; ++j)
#pragma unroll
    for (int i = 0; i < 8; ++i) acc[j][i] = 0.0f;

  for (int k = 0; k < K; k += 32) {
    // ---- async-stage B[n0..n0+63][k..k+31] into LDS ----
    async_copy_b128(B + (size_t)(n0 + r0) * K + k + c0, &bt[r0][c0]);
    async_copy_b128(B + (size_t)(n0 + r1) * K + k + c1, &bt[r1][c1]);
    asm volatile("s_wait_asynccnt 0x0" ::: "memory");
    __syncthreads();

    v16bf a = load_frag(A, K, m0, k);
#pragma unroll
    for (int j = 0; j < 4; ++j) {
      v16bf b = lds_frag(&bt[j * 16][0]);
      acc[j] = __builtin_amdgcn_wmma_f32_16x16x32_bf16(false, a, false, b, (short)0, acc[j],
                                                       false, false);
    }
    __syncthreads();
  }

#pragma unroll
  for (int j = 0; j < 4; ++j) {
    int col = n0 + j * 16 + (lane & 15);
    float bv = (bias && col < Nb) ? bias[col] : 0.0f;
#pragma unroll
    for (int r = 0; r < 8; ++r) {
      int row = m0 + r + 8 * hi;
      float v = acc[j][r] + bv;
      size_t idx = (size_t)row * N + col;
      if (resid) v += resid[idx];
      if (relu)  v = fmaxf(v, 0.0f);
      if (outF) outF[idx] = v;
      if (outB) outB[idx] = f2bf(v);
    }
  }
}

// Flash attention: grid (SP/64, NHEADS), 4 waves/block, each wave = 16 query rows.
__global__ void attention_kernel(const ushort_t* __restrict__ q,
                                 const ushort_t* __restrict__ kk,
                                 const ushort_t* __restrict__ vT,   // [DIMM x SP]
                                 ushort_t* __restrict__ o) {
  __shared__ __align__(16) ushort_t ptile[4][16][32];
  int wid  = threadIdx.x >> 5;
  int lane = threadIdx.x & 31;
  int hi   = lane >> 4;
  int n    = lane & 15;
  int m0   = (blockIdx.x * 4 + wid) * 16;
  int head = blockIdx.y;
  const float SC = 0.17677669529663687f; // 1/sqrt(32)

  v16bf aq = load_frag(q, DIMM, m0, head * DH);
  v8f acc0, acc1, zero8;
#pragma unroll
  for (int i = 0; i < 8; ++i) { acc0[i] = 0.f; acc1[i] = 0.f; zero8[i] = 0.f; }
  float rm[8], rl[8];
#pragma unroll
  for (int r = 0; r < 8; ++r) { rm[r] = -1e30f; rl[r] = 0.f; }

  for (int t0 = 0; t0 < SP; t0 += 32) {
    v16bf bk0 = load_frag(kk, DIMM, t0,      head * DH);
    v16bf bk1 = load_frag(kk, DIMM, t0 + 16, head * DH);
    v8f s0 = __builtin_amdgcn_wmma_f32_16x16x32_bf16(false, aq, false, bk0, (short)0, zero8, false, false);
    v8f s1 = __builtin_amdgcn_wmma_f32_16x16x32_bf16(false, aq, false, bk1, (short)0, zero8, false, false);
#pragma unroll
    for (int r = 0; r < 8; ++r) {
      float e0 = s0[r] * SC; if (t0 + n      >= SVALID) e0 = -1e30f;
      float e1 = s1[r] * SC; if (t0 + 16 + n >= SVALID) e1 = -1e30f;
      float tm = fmaxf(e0, e1);
#pragma unroll
      for (int off = 1; off < 16; off <<= 1) tm = fmaxf(tm, __shfl_xor(tm, off, 32));
      float nm  = fmaxf(rm[r], tm);
      float cor = __expf(rm[r] - nm);
      e0 = __expf(e0 - nm);
      e1 = __expf(e1 - nm);
      float ts = e0 + e1;
#pragma unroll
      for (int off = 1; off < 16; off <<= 1) ts += __shfl_xor(ts, off, 32);
      rl[r] = rl[r] * cor + ts;
      rm[r] = nm;
      acc0[r] *= cor; acc1[r] *= cor;
      int Mr = r + 8 * hi;
      ptile[wid][Mr][n]      = f2bf(e0);
      ptile[wid][Mr][16 + n] = f2bf(e1);
    }
    __syncthreads();
    v16bf ap = lds_frag(&ptile[wid][0][0]);
    v16bf bv0 = load_frag(vT, SP, head * DH,      t0);
    v16bf bv1 = load_frag(vT, SP, head * DH + 16, t0);
    acc0 = __builtin_amdgcn_wmma_f32_16x16x32_bf16(false, ap, false, bv0, (short)0, acc0, false, false);
    acc1 = __builtin_amdgcn_wmma_f32_16x16x32_bf16(false, ap, false, bv1, (short)0, acc1, false, false);
    __syncthreads();
  }
#pragma unroll
  for (int r = 0; r < 8; ++r) {
    int row = m0 + r + 8 * hi;
    float inv = 1.0f / rl[r];
    o[(size_t)row * DIMM + head * DH + n]      = f2bf(acc0[r] * inv);
    o[(size_t)row * DIMM + head * DH + 16 + n] = f2bf(acc1[r] * inv);
  }
}

// rational-quadratic spline + output assembly (pos copy, column 2 rewritten)
__global__ void rqs_kernel(const float* __restrict__ pos, const float* __restrict__ P,
                           float* __restrict__ out, float* __restrict__ ldv) {
  int row = blockIdx.x * blockDim.x + threadIdx.x;
  if (row >= SP) return;
  out[row * 3 + 0] = pos[row * 3 + 0];
  out[row * 3 + 1] = pos[row * 3 + 1];
  out[row * 3 + 2] = pos[row * 3 + 2];
  if (row == 0) return;
  int i = row - 1;
  const float* p = P + (size_t)i * AFFN;
  const float MINB = 1e-4f, MINS = 1e-4f;

  float w[16], hgt[16], dd[17];
  float mx = -1e30f; for (int j = 0; j < 16; ++j) mx = fmaxf(mx, p[j]);
  float sum = 0.f;   for (int j = 0; j < 16; ++j) { w[j] = __expf(p[j] - mx); sum += w[j]; }
  float scl = (1.0f - 16.0f * MINB) / sum;
  for (int j = 0; j < 16; ++j) w[j] = w[j] * scl + MINB;

  mx = -1e30f; for (int j = 0; j < 16; ++j) mx = fmaxf(mx, p[16 + j]);
  sum = 0.f;   for (int j = 0; j < 16; ++j) { hgt[j] = __expf(p[16 + j] - mx); sum += hgt[j]; }
  scl = (1.0f - 16.0f * MINB) / sum;
  for (int j = 0; j < 16; ++j) hgt[j] = hgt[j] * scl + MINB;

  float OFF = logf(expm1f(1.0f - MINS));
  for (int j = 0; j < 17; ++j) {
    float u  = ((j < 16) ? p[32 + j] : p[32]) + OFF;
    float sp = (u > 20.f) ? u : log1pf(__expf(u));
    dd[j] = sp + MINS;
  }

  float x = pos[row * 3 + 2];
  int inside = (x >= 0.f) && (x <= 1.f);
  float xc = fminf(fmaxf(x, 0.f), 1.f);
  float xpos[17], ypos[17]; xpos[0] = 0.f; ypos[0] = 0.f;
  for (int j = 0; j < 16; ++j) { xpos[j + 1] = xpos[j] + w[j]; ypos[j + 1] = ypos[j] + hgt[j]; }
  int k = 0; for (int j = 1; j <= 16; ++j) if (xc >= xpos[j]) k = j;
  if (k > 15) k = 15;

  float xk = xpos[k], xk1 = xpos[k + 1], yk = ypos[k], yk1 = ypos[k + 1];
  float dk = dd[k], dk1 = dd[k + 1];
  float W = xk1 - xk, H = yk1 - yk;
  float s = H / W, z = (xc - xk) / W, z1 = 1.f - z;
  float den = s + (dk1 + dk - 2.f * s) * z * z1;
  float y = yk + H * (s * z * z + dk * z * z1) / den;
  float logdet = 2.f * logf(s) + logf(dk1 * z * z + 2.f * s * z * z1 + dk * z1 * z1) - 2.f * logf(den);

  out[row * 3 + 2] = inside ? y : x;
  ldv[i] = inside ? logdet : 0.f;
}

__global__ void reduce_kernel(const float* __restrict__ ldv, float* __restrict__ out) {
  __shared__ float sm[256];
  int tid = threadIdx.x;
  float s = 0.f;
  for (int j = tid; j < SVALID; j += 256) s += ldv[j];
  sm[tid] = s; __syncthreads();
  for (int st = 128; st > 0; st >>= 1) { if (tid < st) sm[tid] += sm[tid + st]; __syncthreads(); }
  if (tid == 0) *out = sm[0];
}

// ---------------- host ----------------

extern "C" void kernel_launch(void* const* d_in, const int* in_sizes, int n_in,
                              void* d_out, int out_size, void* d_ws, size_t ws_size,
                              hipStream_t stream) {
  (void)in_sizes; (void)n_in; (void)out_size; (void)ws_size;
  const float* pos   = (const float*)d_in[0];
  const float* scale = (const float*)d_in[1];
  const float* press = (const float*)d_in[2];
  const float* temp  = (const float*)d_in[3];
  const float* e_w0  = (const float*)d_in[4];  const float* e_b0 = (const float*)d_in[5];
  const float* e_w1  = (const float*)d_in[6];  const float* e_b1 = (const float*)d_in[7];
  const float* e_w2  = (const float*)d_in[8];  const float* e_b2 = (const float*)d_in[9];
  const float* Wq    = (const float*)d_in[10];
  const float* Wk    = (const float*)d_in[11]; const float* bk   = (const float*)d_in[12];
  const float* Wv    = (const float*)d_in[13];
  const float* Wo    = (const float*)d_in[14]; const float* bo   = (const float*)d_in[15];
  const float* mw0   = (const float*)d_in[16]; const float* mb0  = (const float*)d_in[17];
  const float* mw1   = (const float*)d_in[18]; const float* mb1  = (const float*)d_in[19];
  const float* mw2   = (const float*)d_in[20]; const float* mb2  = (const float*)d_in[21];
  const float* a_w0  = (const float*)d_in[22]; const float* a_b0 = (const float*)d_in[23];
  const float* a_w1  = (const float*)d_in[24]; const float* a_b1 = (const float*)d_in[25];
  const float* a_w2  = (const float*)d_in[26]; const float* a_b2 = (const float*)d_in[27];

  char* ws = (char*)d_ws;
  size_t off = 0;
  auto alloc = [&](size_t bytes) -> void* {
    void* p = ws + off;
    off = (off + bytes + 255) & ~(size_t)255;
    return p;
  };

  // bf16 weight buffers
  ushort_t* w_e0 = (ushort_t*)alloc((size_t)512 * EMBK * 2);
  ushort_t* w_e1 = (ushort_t*)alloc((size_t)512 * 512 * 2);
  ushort_t* w_e2 = (ushort_t*)alloc((size_t)256 * 512 * 2);
  ushort_t* w_q  = (ushort_t*)alloc((size_t)4 * 256 * 256 * 2);
  ushort_t* w_k  = (ushort_t*)alloc((size_t)4 * 256 * 256 * 2);
  ushort_t* w_v  = (ushort_t*)alloc((size_t)4 * 256 * 256 * 2);
  ushort_t* w_o  = (ushort_t*)alloc((size_t)4 * 256 * 256 * 2);
  ushort_t* w_m0 = (ushort_t*)alloc((size_t)4 * 512 * 256 * 2);
  ushort_t* w_m1 = (ushort_t*)alloc((size_t)4 * 512 * 512 * 2);
  ushort_t* w_m2 = (ushort_t*)alloc((size_t)4 * 256 * 512 * 2);
  ushort_t* w_a0 = (ushort_t*)alloc((size_t)512 * 256 * 2);
  ushort_t* w_a1 = (ushort_t*)alloc((size_t)512 * 512 * 2);
  ushort_t* w_a2 = (ushort_t*)alloc((size_t)AFFN * 512 * 2);
  // activations
  ushort_t* X0  = (ushort_t*)alloc((size_t)SP * EMBK * 2);
  float*    hF  = (float*)   alloc((size_t)SP * DIMM * 4);
  ushort_t* hB  = (ushort_t*)alloc((size_t)SP * DIMM * 2);
  ushort_t* t0b = (ushort_t*)alloc((size_t)SP * HIDD * 2);
  ushort_t* t1b = (ushort_t*)alloc((size_t)SP * HIDD * 2);
  ushort_t* qb  = (ushort_t*)alloc((size_t)SP * DIMM * 2);
  ushort_t* kb  = (ushort_t*)alloc((size_t)SP * DIMM * 2);
  ushort_t* vb  = (ushort_t*)alloc((size_t)SP * DIMM * 2);
  ushort_t* vTb = (ushort_t*)alloc((size_t)DIMM * SP * 2);
  ushort_t* ob  = (ushort_t*)alloc((size_t)SP * DIMM * 2);
  float*    par = (float*)   alloc((size_t)SP * AFFN * 4);
  float*    ldv = (float*)   alloc((size_t)SP * 4);

  auto conv = [&](const float* src, ushort_t* dst, int Nin, int Kin, int Np, int Kp) {
    int tot = Np * Kp;
    convert_pad_kernel<<<(tot + 255) / 256, 256, 0, stream>>>(src, dst, Nin, Kin, Np, Kp);
  };
  auto gemm = [&](const ushort_t* A, const ushort_t* B, const float* bias, int Nb,
                  const float* resid, float* oF, ushort_t* oB, int N, int K, int relu) {
    dim3 g(SP / 64, (N + 63) / 64);
    gemm_bf16_kernel<<<g, 128, 0, stream>>>(A, B, bias, Nb, resid, oF, oB, N, K, relu);
  };

  // weight conversion to bf16 (with K padding for the 35-wide embedding input)
  conv(e_w0, w_e0,  512,  35,  512, EMBK);
  conv(e_w1, w_e1,  512, 512,  512, 512);
  conv(e_w2, w_e2,  256, 512,  256, 512);
  conv(Wq,   w_q,  1024, 256, 1024, 256);
  conv(Wk,   w_k,  1024, 256, 1024, 256);
  conv(Wv,   w_v,  1024, 256, 1024, 256);
  conv(Wo,   w_o,  1024, 256, 1024, 256);
  conv(mw0,  w_m0, 2048, 256, 2048, 256);
  conv(mw1,  w_m1, 2048, 512, 2048, 512);
  conv(mw2,  w_m2, 1024, 512, 1024, 512);
  conv(a_w0, w_a0,  512, 256,  512, 256);
  conv(a_w1, w_a1,  512, 512,  512, 512);
  conv(a_w2, w_a2,   49, 512, AFFN, 512);

  // conditioning features
  cond_kernel<<<(SP + 127) / 128, 128, 0, stream>>>(pos, scale, press, temp, X0);

  // embedding MLP: 64 -> 512 -> 512 -> 256 (residual stream hF/hB)
  gemm(X0,  w_e0, e_b0, HIDD, nullptr, nullptr, t0b, HIDD, EMBK, 1);
  gemm(t0b, w_e1, e_b1, HIDD, nullptr, nullptr, t1b, HIDD, HIDD, 1);
  gemm(t1b, w_e2, e_b2, DIMM, nullptr, hF, hB, DIMM, HIDD, 0);

  for (int b = 0; b < 4; ++b) {
    const ushort_t* wqb = w_q  + (size_t)b * 256 * 256;
    const ushort_t* wkb = w_k  + (size_t)b * 256 * 256;
    const ushort_t* wvb = w_v  + (size_t)b * 256 * 256;
    const ushort_t* wob = w_o  + (size_t)b * 256 * 256;
    const ushort_t* m0b = w_m0 + (size_t)b * 512 * 256;
    const ushort_t* m1b = w_m1 + (size_t)b * 512 * 512;
    const ushort_t* m2b = w_m2 + (size_t)b * 256 * 512;

    gemm(hB, wqb, nullptr, 0,        nullptr, nullptr, qb, DIMM, DIMM, 0);
    gemm(hB, wkb, bk + b * 256, DIMM, nullptr, nullptr, kb, DIMM, DIMM, 0);
    gemm(hB, wvb, nullptr, 0,        nullptr, nullptr, vb, DIMM, DIMM, 0);
    transpose_kernel<<<(SP * DIMM + 255) / 256, 256, 0, stream>>>(vb, vTb);
    attention_kernel<<<dim3(SP / 64, NHEADS), 128, 0, stream>>>(qb, kb, vTb, ob);
    gemm(ob, wob, bo + b * 256, DIMM, hF, hF, hB, DIMM, DIMM, 0);   // h += attn_out

    gemm(hB,  m0b, mb0 + b * 512, HIDD, nullptr, nullptr, t0b, HIDD, DIMM, 1);
    gemm(t0b, m1b, mb1 + b * 512, HIDD, nullptr, nullptr, t1b, HIDD, HIDD, 1);
    gemm(t1b, m2b, mb2 + b * 256, DIMM, hF, hF, hB, DIMM, HIDD, 0); // h += mlp_out
  }

  // head MLP: 256 -> 512 -> 512 -> 49 (padded to 64)
  gemm(hB,  w_a0, a_b0, HIDD, nullptr, nullptr, t0b, HIDD, DIMM, 1);
  gemm(t0b, w_a1, a_b1, HIDD, nullptr, nullptr, t1b, HIDD, HIDD, 1);
  gemm(t1b, w_a2, a_b2, 49,   nullptr, par, nullptr, AFFN, HIDD, 0);

  // spline transform + output assembly, then deterministic logdet reduction
  rqs_kernel<<<(SP + 127) / 128, 128, 0, stream>>>(pos, par, (float*)d_out, ldv);
  reduce_kernel<<<1, 256, 0, stream>>>(ldv, (float*)d_out + SP * 3);
}
